// AttentionHead_17128329576952
// MI455X (gfx1250) — matrix operand order repaired and checked by reading, not compiled, not run on hardware
//
#include <hip/hip_runtime.h>
#include <hip/hip_bf16.h>

typedef __bf16 bf16_t;
typedef __attribute__((ext_vector_type(16))) __bf16 v16bf;
typedef __attribute__((ext_vector_type(8)))  __bf16 v8bf;
typedef __attribute__((ext_vector_type(8)))  float  v8f;

#define BATCH 4
#define NTOK  2048
#define EDIM  1024
#define HDIM  64

// gfx1250 has native bf16: let the compiler emit hardware cvt (RNE).
static __device__ inline bf16_t f2bf(float f) { return (bf16_t)f; }

#define WMMA_BF16(A, B, C) \
  __builtin_amdgcn_wmma_f32_16x16x32_bf16(false, (A), false, (B), (short)0, (C), false, false)

// ---------------------------------------------------------------------------
// Kernel 1: transpose + downcast weights: WT[n][k] = (bf16)W[k][n]
// Makes the WMMA B-operand load (fixed N per lane, contiguous K) one v16bf.
// ---------------------------------------------------------------------------
__global__ void prep_weights(const float* __restrict__ Wk,
                             const float* __restrict__ Wq,
                             const float* __restrict__ Wv,
                             bf16_t* __restrict__ WqT,
                             bf16_t* __restrict__ WkT,
                             bf16_t* __restrict__ WvT) {
  int t = blockIdx.x * blockDim.x + threadIdx.x;   // 0 .. 65535
  const float* W; bf16_t* WT;
  if (blockIdx.y == 0)      { W = Wq; WT = WqT; }
  else if (blockIdx.y == 1) { W = Wk; WT = WkT; }
  else                      { W = Wv; WT = WvT; }
  int n = t & (HDIM - 1);
  int k = t >> 6;
  WT[(size_t)n * EDIM + k] = f2bf(W[(size_t)k * HDIM + n]);
}

// ---------------------------------------------------------------------------
// Kernel 2: projections  q = Q@Wq, k = K@Wk, v = V@Wv  via bf16 WMMA,
// f32 accumulate.  q,k stored row-major bf16 [B][N][64]; v stored transposed
// [B][64][N] so the attention P·V B-operand is two contiguous v8bf loads.
// One wave = one 16-row tile x 4 col tiles; K loop 1024/32 -> 128 WMMAs.
// ---------------------------------------------------------------------------
__global__ void proj_kernel(const float* __restrict__ Q,
                            const float* __restrict__ K,
                            const float* __restrict__ V,
                            const bf16_t* __restrict__ WqT,
                            const bf16_t* __restrict__ WkT,
                            const bf16_t* __restrict__ WvT,
                            bf16_t* __restrict__ qO,
                            bf16_t* __restrict__ kO,
                            bf16_t* __restrict__ vTO) {
  const int lane = threadIdx.x & 31;
  const int wave = threadIdx.x >> 5;
  const int tile = blockIdx.x * 4 + wave;          // 0..511 (8192 rows / 16)
  const int which = blockIdx.y;                    // 0:q 1:k 2:v

  const float* X; const bf16_t* WT;
  if (which == 0)      { X = Q; WT = WqT; }
  else if (which == 1) { X = K; WT = WkT; }
  else                 { X = V; WT = WvT; }

  const int n    = lane & 15;
  const int half = lane >> 4;

  v8f acc0 = {}, acc1 = {}, acc2 = {}, acc3 = {};

  // 16-bit A 16x32 layout: lane holds row (lane&15);
  // halves 0..15 -> K = half*16+0..15 (contiguous, matches B)
  const float* xrow = X + (size_t)(tile * 16 + n) * EDIM;

  for (int kb = 0; kb < EDIM; kb += 32) {
    v8f f0 = *(const v8f*)(xrow + kb + half * 16);
    v8f f1 = *(const v8f*)(xrow + kb + half * 16 + 8);
    v16bf a = {};
#pragma unroll
    for (int i = 0; i < 8; ++i) { a[i] = f2bf(f0[i]); a[i + 8] = f2bf(f1[i]); }

    // B-operand: lane holds column n, halves h -> K = half*16 + h (contiguous)
    const bf16_t* wb = WT + kb + half * 16;
    v16bf b0 = *(const v16bf*)(wb + (size_t)(0 * 16 + n) * EDIM);
    v16bf b1 = *(const v16bf*)(wb + (size_t)(1 * 16 + n) * EDIM);
    v16bf b2 = *(const v16bf*)(wb + (size_t)(2 * 16 + n) * EDIM);
    v16bf b3 = *(const v16bf*)(wb + (size_t)(3 * 16 + n) * EDIM);

    acc0 = WMMA_BF16(a, b0, acc0);
    acc1 = WMMA_BF16(a, b1, acc1);
    acc2 = WMMA_BF16(a, b2, acc2);
    acc3 = WMMA_BF16(a, b3, acc3);
  }

  // C/D layout: VGPR r holds (M = r + half*8, N = n)
#pragma unroll
  for (int r = 0; r < 8; ++r) {
    int rowIdx = tile * 16 + r + half * 8;
    if (which < 2) {
      bf16_t* dst = (which == 0) ? qO : kO;
      size_t base = (size_t)rowIdx * HDIM + n;
      dst[base + 0]  = f2bf(acc0[r]);
      dst[base + 16] = f2bf(acc1[r]);
      dst[base + 32] = f2bf(acc2[r]);
      dst[base + 48] = f2bf(acc3[r]);
    } else {
      int bb  = rowIdx >> 11;
      int tok = rowIdx & (NTOK - 1);
      bf16_t* dst = vTO + (size_t)bb * HDIM * NTOK + tok;
      dst[(size_t)(n + 0)  * NTOK] = f2bf(acc0[r]);
      dst[(size_t)(n + 16) * NTOK] = f2bf(acc1[r]);
      dst[(size_t)(n + 32) * NTOK] = f2bf(acc2[r]);
      dst[(size_t)(n + 48) * NTOK] = f2bf(acc3[r]);
    }
  }
}

// ---------------------------------------------------------------------------
// Kernel 3: causal flash attention, one wave per 16-query tile.
// Trick: compute S^T = K_tile x Q^T so the exp'd probabilities land in the
// exact lane-local layout needed for the P-as-A-operand of the P·V WMMA --
// no LDS transpose, and softmax row stats are per-lane scalars with a single
// xor-16 shuffle to combine the two key-halves.
// ---------------------------------------------------------------------------
__global__ void attn_kernel(const bf16_t* __restrict__ qB,
                            const bf16_t* __restrict__ kB,
                            const bf16_t* __restrict__ vT,
                            float* __restrict__ out) {
  const int lane = threadIdx.x & 31;
  const int wave = threadIdx.x >> 5;
  const int qt   = blockIdx.x * 4 + wave;          // 0..127
  const int b    = blockIdx.y;
  const int n    = lane & 15;
  const int half = lane >> 4;

  // q^T as WMMA B-operands (loop invariant): lane = query n,
  // halves h -> d = half*16 + h (chunk 0), 32 + half*16 + h (chunk 1)
  const bf16_t* qrow = qB + ((size_t)b * NTOK + qt * 16 + n) * HDIM + half * 16;
  v16bf qb0 = *(const v16bf*)(qrow);
  v16bf qb1 = *(const v16bf*)(qrow + 32);

  v8f O0 = {}, O1 = {}, O2 = {}, O3 = {};
  float mrun = -__builtin_inff();
  float lrun = 0.0f;

  const int qtok = qt * 16 + n;        // this lane's query token
  const int nblk = ((qt * 16 + 15) >> 5) + 1;

  for (int kblk = 0; kblk < nblk; ++kblk) {
    const int keyBase = kblk * 32;

    if (kblk + 1 < nblk)
      __builtin_prefetch(kB + ((size_t)b * NTOK + keyBase + 32) * HDIM, 0, 0);

    // ---- A operands: two 16-key tiles of k (16 x 32 each over d)
    const bf16_t* krow0 = kB + ((size_t)b * NTOK + keyBase + n) * HDIM;
    const bf16_t* krow1 = krow0 + (size_t)16 * HDIM;
    v16bf ka00 = {}, ka01 = {}, ka10 = {}, ka11 = {};
    {
      v8bf t0 = *(const v8bf*)(krow0 + half * 16);
      v8bf t1 = *(const v8bf*)(krow0 + 8 + half * 16);
      v8bf t2 = *(const v8bf*)(krow0 + 32 + half * 16);
      v8bf t3 = *(const v8bf*)(krow0 + 40 + half * 16);
      v8bf u0 = *(const v8bf*)(krow1 + half * 16);
      v8bf u1 = *(const v8bf*)(krow1 + 8 + half * 16);
      v8bf u2 = *(const v8bf*)(krow1 + 32 + half * 16);
      v8bf u3 = *(const v8bf*)(krow1 + 40 + half * 16);
#pragma unroll
      for (int i = 0; i < 8; ++i) {
        ka00[i] = t0[i]; ka00[i + 8] = t1[i];
        ka01[i] = t2[i]; ka01[i + 8] = t3[i];
        ka10[i] = u0[i]; ka10[i + 8] = u1[i];
        ka11[i] = u2[i]; ka11[i + 8] = u3[i];
      }
    }

    // ---- S^T = k . q^T : C-layout element (key = r + half*8, query = n)
    v8f s0 = {}, s1 = {};
    s0 = WMMA_BF16(ka00, qb0, s0);
    s0 = WMMA_BF16(ka01, qb1, s0);
    s1 = WMMA_BF16(ka10, qb0, s1);
    s1 = WMMA_BF16(ka11, qb1, s1);

    // ---- causal mask + online softmax; each lane owns 16 of one query's 32 scores
    const float sc = 0.125f;             // 1/sqrt(64)
    float p0[8], p1[8];
    float vmax = -__builtin_inff();
#pragma unroll
    for (int r = 0; r < 8; ++r) {
      int key = keyBase + r + half * 8;
      float a0 = (key      <= qtok) ? s0[r] * sc : -__builtin_inff();
      float a1 = (key + 16 <= qtok) ? s1[r] * sc : -__builtin_inff();
      p0[r] = a0; p1[r] = a1;
      vmax = fmaxf(vmax, fmaxf(a0, a1));
    }
    vmax = fmaxf(vmax, __shfl_xor(vmax, 16, 32));   // combine key-halves

    float mnew = fmaxf(mrun, vmax);
    float corr = __expf(mrun - mnew);
    float psum = 0.0f;
#pragma unroll
    for (int r = 0; r < 8; ++r) {
      p0[r] = __expf(p0[r] - mnew);
      p1[r] = __expf(p1[r] - mnew);
      psum += p0[r] + p1[r];
    }
    psum += __shfl_xor(psum, 16, 32);
    lrun = lrun * corr + psum;
    mrun = mnew;

    // ---- rescale O (C-layout row = query r + half*8) by that query's corr
#pragma unroll
    for (int r = 0; r < 8; ++r) {
      float cf = __shfl(corr, r + half * 8, 32);
      O0[r] *= cf; O1[r] *= cf; O2[r] *= cf; O3[r] *= cf;
    }

    // ---- P as A-operand: entirely lane-local (the S^T trick)
    v16bf pa = {};
#pragma unroll
    for (int h = 0; h < 8; ++h) { pa[h] = f2bf(p0[h]); pa[h + 8] = f2bf(p1[h]); }

    // ---- O += P @ V ; vT is [B][64][N]; B-operand keys 8h+0..7 and 16+8h+0..7 (matches P)
    const bf16_t* vrow = vT + ((size_t)b * HDIM + n) * NTOK + keyBase + half * 8;
    v16bf vb0 = __builtin_shufflevector(*(const v8bf*)(vrow), *(const v8bf*)(vrow + 16), 0, 1, 2, 3, 4, 5, 6, 7, 8, 9, 10, 11, 12, 13, 14, 15);
    v16bf vb1 = __builtin_shufflevector(*(const v8bf*)(vrow + (size_t)16 * NTOK), *(const v8bf*)(vrow + (size_t)16 * NTOK + 16), 0, 1, 2, 3, 4, 5, 6, 7, 8, 9, 10, 11, 12, 13, 14, 15);
    v16bf vb2 = __builtin_shufflevector(*(const v8bf*)(vrow + (size_t)32 * NTOK), *(const v8bf*)(vrow + (size_t)32 * NTOK + 16), 0, 1, 2, 3, 4, 5, 6, 7, 8, 9, 10, 11, 12, 13, 14, 15);
    v16bf vb3 = __builtin_shufflevector(*(const v8bf*)(vrow + (size_t)48 * NTOK), *(const v8bf*)(vrow + (size_t)48 * NTOK + 16), 0, 1, 2, 3, 4, 5, 6, 7, 8, 9, 10, 11, 12, 13, 14, 15);
    O0 = WMMA_BF16(pa, vb0, O0);
    O1 = WMMA_BF16(pa, vb1, O1);
    O2 = WMMA_BF16(pa, vb2, O2);
    O3 = WMMA_BF16(pa, vb3, O3);
  }

  // ---- finalize: divide row (query) by its softmax denominator, write fp32
  float linv = 1.0f / lrun;
#pragma unroll
  for (int r = 0; r < 8; ++r) {
    float lf = __shfl(linv, r + half * 8, 32);
    size_t base = ((size_t)b * NTOK + qt * 16 + r + half * 8) * HDIM + n;
    out[base + 0]  = O0[r] * lf;
    out[base + 16] = O1[r] * lf;
    out[base + 32] = O2[r] * lf;
    out[base + 48] = O3[r] * lf;
  }
}

// ---------------------------------------------------------------------------
extern "C" void kernel_launch(void* const* d_in, const int* in_sizes, int n_in,
                              void* d_out, int out_size, void* d_ws, size_t ws_size,
                              hipStream_t stream) {
  (void)in_sizes; (void)n_in; (void)out_size; (void)ws_size;

  // setup_inputs order: K, Q, V, Wk, Wq, Wv
  const float* K  = (const float*)d_in[0];
  const float* Q  = (const float*)d_in[1];
  const float* V  = (const float*)d_in[2];
  const float* Wk = (const float*)d_in[3];
  const float* Wq = (const float*)d_in[4];
  const float* Wv = (const float*)d_in[5];
  float* out = (float*)d_out;

  bf16_t* WqT = (bf16_t*)d_ws;                                 // 64*1024 bf16
  bf16_t* WkT = WqT + (size_t)HDIM * EDIM;
  bf16_t* WvT = WkT + (size_t)HDIM * EDIM;
  bf16_t* qB  = WvT + (size_t)HDIM * EDIM;                     // [B][N][64]
  bf16_t* kB  = qB  + (size_t)BATCH * NTOK * HDIM;             // [B][N][64]
  bf16_t* vT  = kB  + (size_t)BATCH * NTOK * HDIM;             // [B][64][N]

  prep_weights<<<dim3(256, 3), 256, 0, stream>>>(Wk, Wq, Wv, WqT, WkT, WvT);
  proj_kernel<<<dim3(128, 3), 128, 0, stream>>>(Q, K, V, WqT, WkT, WvT, qB, kB, vT);
  attn_kernel<<<dim3(32, 4), 128, 0, stream>>>(qB, kB, vT, out);
}
